// S4Model_33028298506609
// MI455X (gfx1250) — matrix operand order repaired and checked by reading, not compiled, not run on hardware
//
#include <hip/hip_runtime.h>

// ---------------- model constants ----------------
#define H_      1024
#define N_      64
#define LAYERS_ 4
#define B_      16
#define L_      2048
#define BT_     (B_*L_)      // 32768 tokens

// ---------------- WMMA vector types ----------------
typedef __bf16 v16bf __attribute__((ext_vector_type(16)));
typedef float  v8f   __attribute__((ext_vector_type(8)));

// f32 -> bf16 (round-to-nearest-even)
__device__ __forceinline__ unsigned short f2bf(float f){
  unsigned int u = __float_as_uint(f);
  u += 0x7FFFu + ((u >> 16) & 1u);
  return (unsigned short)(u >> 16);
}

// ---------------- gfx1250 async global->LDS copy (ASYNCcnt) ----------------
#if defined(__has_builtin)
#if __has_builtin(__builtin_amdgcn_global_load_async_to_lds_b128)
#define HAVE_ASYNC 1
#else
#define HAVE_ASYNC 0
#endif
#else
#define HAVE_ASYNC 0
#endif

// builtin parameter type per hipcc diagnostic: pointer to GCC-vector int4
typedef int v4i __attribute__((vector_size(16)));
typedef __attribute__((address_space(1))) v4i* v4i_gp;   // global
typedef __attribute__((address_space(3))) v4i* v4i_lp;   // LDS

__device__ __forceinline__ void async_cp16(const unsigned short* g, unsigned short* l){
#if HAVE_ASYNC
  __builtin_amdgcn_global_load_async_to_lds_b128((v4i_gp)(g), (v4i_lp)(l), 0, 0);
#else
  *(uint4*)l = *(const uint4*)g;
#endif
}
__device__ __forceinline__ void wait_async0(){
#if HAVE_ASYNC
#if __has_builtin(__builtin_amdgcn_s_wait_asynccnt)
  __builtin_amdgcn_s_wait_asynccnt(0);
#else
  asm volatile("s_wait_asynccnt 0" ::: "memory");
#endif
#endif
}

// ============================================================
// 1) Embedding gather + sinusoidal positional encoding
// ============================================================
__global__ void __launch_bounds__(256)
embed_kernel(const int* __restrict__ x,
             const float* __restrict__ emb,
             float* __restrict__ h){
  int i   = blockIdx.x * blockDim.x + threadIdx.x;   // < BT_*H_ = 2^25
  int c   = i & (H_ - 1);
  int tok = i >> 10;
  int t   = tok & (L_ - 1);
  int v   = x[tok];
  int p   = c & ~1;
  float div = __expf((float)p * (-9.210340371976184f / (float)H_)); // -ln(1e4)/H
  float ang = (float)t * div;
  float pe  = (c & 1) ? __cosf(ang) : __sinf(ang);
  h[i] = emb[(long long)v * H_ + c] + pe;
}

// ============================================================
// 2) S4D scan: one wave32 per (b,h) channel, 2 complex states/lane
// ============================================================
__global__ void __launch_bounds__(256)
s4d_scan_kernel(const float* __restrict__ u,
                float* __restrict__ y,
                const float* __restrict__ log_dt,
                const float* __restrict__ log_A_real,
                const float* __restrict__ A_imag,
                const float* __restrict__ C_real,
                const float* __restrict__ C_imag,
                const float* __restrict__ Dp){
  int wave = (blockIdx.x * blockDim.x + threadIdx.x) >> 5;
  int lane = threadIdx.x & 31;
  int b    = wave >> 10;
  int hh   = wave & (H_ - 1);

  float dt = __expf(log_dt[hh]);
  float dp = Dp[hh];

  float abr[2], abi[2], bbr[2], bbi[2], cr[2], ci[2], sr[2], si[2];
#pragma unroll
  for (int j = 0; j < 2; ++j){
    int   off = hh * N_ + lane + j * 32;
    float are = -__expf(log_A_real[off]);
    float aim = A_imag[off];
    float er  = __expf(are * dt);
    float re  = er * __cosf(aim * dt);
    float im  = er * __sinf(aim * dt);
    abr[j] = re; abi[j] = im;
    float inv = 1.0f / (are * are + aim * aim);
    bbr[j] = ((re - 1.f) * are + im * aim) * inv;
    bbi[j] = (im * are - (re - 1.f) * aim) * inv;
    cr[j] = C_real[off]; ci[j] = C_imag[off];
    sr[j] = 0.f; si[j] = 0.f;
  }

  const float* up = u + ((long long)b * L_) * H_ + hh;
  float*       yp = y + ((long long)b * L_) * H_ + hh;

  float ucur = up[0];
  for (int t = 0; t < L_; ++t){
    float unext = (t + 1 < L_) ? up[(t + 1) * H_] : 0.f;
    float acc = 0.f;
#pragma unroll
    for (int j = 0; j < 2; ++j){
      float nr = abr[j] * sr[j] - abi[j] * si[j] + bbr[j] * ucur;
      float ni = abr[j] * si[j] + abi[j] * sr[j] + bbi[j] * ucur;
      sr[j] = nr; si[j] = ni;
      acc += nr * cr[j] - ni * ci[j];
    }
#pragma unroll
    for (int off = 16; off >= 1; off >>= 1)
      acc += __shfl_xor(acc, off, 32);
    if (lane == 0) yp[t * H_] = 2.f * acc + ucur * dp;
    ucur = unext;
  }
}

// ============================================================
// 3) LayerNorm over H per token -> bf16 output
// ============================================================
__global__ void __launch_bounds__(256)
ln_kernel(const float* __restrict__ y,
          const float* __restrict__ g,
          const float* __restrict__ bb,
          unsigned short* __restrict__ outbf){
  __shared__ float red[16];
  int tid = threadIdx.x;
  const float* row = y + (long long)blockIdx.x * H_;
  unsigned short* orow = outbf + (long long)blockIdx.x * H_;
  float v[4], s = 0.f, ss = 0.f;
#pragma unroll
  for (int k = 0; k < 4; ++k){
    v[k] = row[tid + k * 256];
    s += v[k]; ss += v[k] * v[k];
  }
#pragma unroll
  for (int off = 16; off >= 1; off >>= 1){
    s  += __shfl_xor(s,  off, 32);
    ss += __shfl_xor(ss, off, 32);
  }
  if ((tid & 31) == 0){ red[tid >> 5] = s; red[8 + (tid >> 5)] = ss; }
  __syncthreads();
  if (tid < 32){
    float a  = (tid < 8) ? red[tid]     : 0.f;
    float b2 = (tid < 8) ? red[8 + tid] : 0.f;
#pragma unroll
    for (int off = 4; off >= 1; off >>= 1){
      a  += __shfl_xor(a,  off, 32);
      b2 += __shfl_xor(b2, off, 32);
    }
    if (tid == 0){ red[0] = a; red[1] = b2; }
  }
  __syncthreads();
  float mu  = red[0] * (1.f / H_);
  float var = red[1] * (1.f / H_) - mu * mu;
  float rs  = rsqrtf(var + 1e-5f);
#pragma unroll
  for (int k = 0; k < 4; ++k){
    int c = tid + k * 256;
    orow[c] = f2bf((v[k] - mu) * rs * g[c] + bb[c]);
  }
}

// ============================================================
// 4) Weight transpose+convert: W[K][N] f32 -> Wt[N][K] bf16
// ============================================================
__global__ void __launch_bounds__(256)
wtrans_kernel(const float* __restrict__ W, unsigned short* __restrict__ Wt){
  __shared__ float tile[32][33];
  int bx = blockIdx.x;            // N tile
  int by = blockIdx.y;            // K tile
  int tx = threadIdx.x & 31, ty = threadIdx.x >> 5;
#pragma unroll
  for (int i = 0; i < 32; i += 8)
    tile[ty + i][tx] = W[(long long)(by * 32 + ty + i) * H_ + bx * 32 + tx];
  __syncthreads();
#pragma unroll
  for (int i = 0; i < 32; i += 8)
    Wt[(long long)(bx * 32 + ty + i) * H_ + by * 32 + tx] = f2bf(tile[tx][ty + i]);
}

// ============================================================
// 5) f32 -> bf16 elementwise convert
// ============================================================
__global__ void __launch_bounds__(256)
cvt_bf16_kernel(const float* __restrict__ in, unsigned short* __restrict__ out){
  int i = blockIdx.x * blockDim.x + threadIdx.x;
  out[i] = f2bf(in[i]);
}

// ============================================================
// 6) WMMA bf16 GEMM, async double-buffered LDS staging.
//    In bf16 [M][K] @ Wt bf16 [N][K] (+bias).
//    MODE 0: relu -> bf16 OutBf
//    MODE 1: residual f32 OutF += relu
//    MODE 2: plain  -> f32 OutF
// ============================================================
#define TM   128
#define TN   128
#define TK   32
#define LDSS 40      // ushort row stride (80B): 16B aligned, conflict-padded

template<int MODE>
__global__ void __launch_bounds__(256)
gemm_bf16_kernel(const unsigned short* __restrict__ Abf,
                 const unsigned short* __restrict__ Wt,
                 const float* __restrict__ bias,
                 float* __restrict__ OutF,
                 unsigned short* __restrict__ OutBf){
  __shared__ unsigned short Alds[2][TM * LDSS];
  __shared__ unsigned short Blds[2][TN * LDSS];

  const int tid  = threadIdx.x;
  const int lane = tid & 31;
  const int warp = tid >> 5;
  const int m0 = blockIdx.y * TM;
  const int n0 = blockIdx.x * TN;
  const int wm = (warp >> 1) * 32;
  const int wn = (warp & 1) * 64;
  const int K = H_, Nout = H_;

  v8f acc[2][4];
#pragma unroll
  for (int a = 0; a < 2; ++a)
#pragma unroll
    for (int b = 0; b < 4; ++b)
#pragma unroll
      for (int e = 0; e < 8; ++e) acc[a][b][e] = 0.f;

  // stage one K-slab (A: 128x32 bf16, B: 128x32 bf16) via async b128 copies
  auto stage = [&](int bsel, int kk){
#pragma unroll
    for (int s = 0; s < 2; ++s){
      int slot = tid + s * 256;          // 0..511
      int r    = slot >> 2;              // row 0..127
      int c8   = (slot & 3) << 3;        // k-elem offset 0,8,16,24
      async_cp16(Abf + (long long)(m0 + r) * K + kk + c8,
                 &Alds[bsel][r * LDSS + c8]);
      async_cp16(Wt  + (long long)(n0 + r) * K + kk + c8,
                 &Blds[bsel][r * LDSS + c8]);
    }
  };

  int buf = 0;
  stage(0, 0);
  wait_async0();
  __syncthreads();

  for (int kk = 0; kk < K; kk += TK){
    if (kk + TK < K) stage(buf ^ 1, kk + TK);   // overlap next slab

    // ---- build fragments per ISA 7.12.2 layouts ----
    union Frag { v16bf v; unsigned int u[8]; };
    Frag af[2], bfm[4];
    const int l15 = lane & 15;
    const int akb = (lane < 16) ? 0 : 8;    // A: K 0..7/16..23 vs 8..15/24..31
    const int bkb = (lane < 16) ? 0 : 8;    // B: K 0..7/16..23 vs 8..15/24..31
#pragma unroll
    for (int fm = 0; fm < 2; ++fm){
      const unsigned short* ap = &Alds[buf][(wm + fm * 16 + l15) * LDSS];
#pragma unroll
      for (int i = 0; i < 4; ++i){
        af[fm].u[i]     = *(const unsigned int*)(ap + akb + 2 * i);
        af[fm].u[4 + i] = *(const unsigned int*)(ap + akb + 16 + 2 * i);
      }
    }
#pragma unroll
    for (int fn = 0; fn < 4; ++fn){
      const unsigned short* bp = &Blds[buf][(wn + fn * 16 + l15) * LDSS + bkb];
#pragma unroll
      for (int i = 0; i < 4; ++i){
        bfm[fn].u[i]     = *(const unsigned int*)(bp + 2 * i);
        bfm[fn].u[4 + i] = *(const unsigned int*)(bp + 16 + 2 * i);
      }
    }
#pragma unroll
    for (int fm = 0; fm < 2; ++fm)
#pragma unroll
      for (int fn = 0; fn < 4; ++fn)
        acc[fm][fn] = __builtin_amdgcn_wmma_f32_16x16x32_bf16(
            false, af[fm].v, false, bfm[fn].v,
            (short)0, acc[fm][fn], false, false);

    wait_async0();      // our next-slab copies have landed
    __syncthreads();    // everyone done reading buf / writing buf^1
    buf ^= 1;
  }

  // ---- epilogue (C/D: VGPR v -> M=v (+8 upper half-wave), N=lane&15) ----
  const int l15  = lane & 15;
  const int rsel = (lane < 16) ? 0 : 8;
#pragma unroll
  for (int fm = 0; fm < 2; ++fm){
#pragma unroll
    for (int fn = 0; fn < 4; ++fn){
      int col  = n0 + wn + fn * 16 + l15;
      float bv = bias[col];
#pragma unroll
      for (int v = 0; v < 8; ++v){
        int row = m0 + wm + fm * 16 + rsel + v;
        float val = acc[fm][fn][v] + bv;
        long long idx = (long long)row * Nout + col;
        if (MODE == 0){
          OutBf[idx] = f2bf(fmaxf(val, 0.f));
        } else if (MODE == 1){
          OutF[idx] = OutF[idx] + fmaxf(val, 0.f);
        } else {
          OutF[idx] = val;
        }
      }
    }
  }
}

// ============================================================
// host-side launcher
// ============================================================
extern "C" void kernel_launch(void* const* d_in, const int* in_sizes, int n_in,
                              void* d_out, int out_size, void* d_ws, size_t ws_size,
                              hipStream_t stream){
  (void)in_sizes; (void)n_in; (void)out_size; (void)ws_size;
  const int*   x          = (const int*)  d_in[0];
  const float* emb        = (const float*)d_in[1];
  const float* log_dt     = (const float*)d_in[2];
  const float* log_A_real = (const float*)d_in[3];
  const float* A_imag     = (const float*)d_in[4];
  const float* C_real     = (const float*)d_in[5];
  const float* C_imag     = (const float*)d_in[6];
  const float* Dp         = (const float*)d_in[7];
  const float* ln_g       = (const float*)d_in[8];
  const float* ln_b       = (const float*)d_in[9];
  const float* W1         = (const float*)d_in[10];
  const float* b1         = (const float*)d_in[11];
  const float* W2         = (const float*)d_in[12];
  const float* b2         = (const float*)d_in[13];
  const float* oW         = (const float*)d_in[14];
  const float* ob         = (const float*)d_in[15];
  float* out = (float*)d_out;

  const size_t TOK = (size_t)BT_ * H_;
  const size_t HH  = (size_t)H_ * H_;
  float*          hbuf = (float*)d_ws;                 // residual stream (f32)
  float*          ybuf = hbuf + TOK;                   // scan output (f32)
  unsigned short* abf  = (unsigned short*)(ybuf + TOK);// bf16 activations
  unsigned short* zbf  = abf + TOK;                    // bf16 MLP hidden
  unsigned short* wt   = zbf + TOK;                    // 9 transposed bf16 weights

  // one-time (per launch) weight transpose+convert: [K][N]f32 -> [N][K]bf16
  dim3 tgrid(H_ / 32, H_ / 32);
  for (int l = 0; l < LAYERS_; ++l){
    wtrans_kernel<<<tgrid, 256, 0, stream>>>(W1 + l * HH, wt + (size_t)(2 * l) * HH);
    wtrans_kernel<<<tgrid, 256, 0, stream>>>(W2 + l * HH, wt + (size_t)(2 * l + 1) * HH);
  }
  wtrans_kernel<<<tgrid, 256, 0, stream>>>(oW, wt + (size_t)8 * HH);

  embed_kernel<<<(BT_ * H_) / 256, 256, 0, stream>>>(x, emb, hbuf);

  dim3 ggrid(H_ / TN, BT_ / TM);    // (8, 256)
  for (int l = 0; l < LAYERS_; ++l){
    const size_t po = (size_t)l * H_ * N_;
    s4d_scan_kernel<<<(B_ * H_) / 8, 256, 0, stream>>>(
        hbuf, ybuf, log_dt + l * H_, log_A_real + po, A_imag + po,
        C_real + po, C_imag + po, Dp + l * H_);
    ln_kernel<<<BT_, 256, 0, stream>>>(ybuf, ln_g + l * H_, ln_b + l * H_, abf);
    gemm_bf16_kernel<0><<<ggrid, 256, 0, stream>>>(abf, wt + (size_t)(2 * l) * HH,
                                                   b1 + l * H_, nullptr, zbf);
    gemm_bf16_kernel<1><<<ggrid, 256, 0, stream>>>(zbf, wt + (size_t)(2 * l + 1) * HH,
                                                   b2 + l * H_, hbuf, nullptr);
  }
  cvt_bf16_kernel<<<(BT_ * H_) / 256, 256, 0, stream>>>(hbuf, abf);
  gemm_bf16_kernel<2><<<ggrid, 256, 0, stream>>>(abf, wt + (size_t)8 * HH,
                                                 ob, out, nullptr);
}
